// CronRootAttentionHybrid_35759897707056
// MI455X (gfx1250) — compile-verified, matrix operand order checked by reading, not run on hardware
//
#include <hip/hip_runtime.h>
#include <hip/hip_bf16.h>

// ---------------------------------------------------------------------------
// CronRootAttentionHybrid for MI455X (gfx1250, wave32, WMMA, TDM)
//
// Pipeline:
//   [0] f32 -> f16 conversion of x and the four weight matrices
//   [1] fused QKV projection GEMM  (wave-per-16x16-tile, v_wmma_f32_16x16x32_f16)
//   [2] flash-attention (causal, online softmax, LDS P re-layout)
//   [3] output projection GEMM: TDM (tensor_load_to_lds) staged LDS tiles -> f32
// ---------------------------------------------------------------------------

typedef __attribute__((ext_vector_type(16))) _Float16 v16h;
typedef __attribute__((ext_vector_type(8)))  _Float16 v8h;
typedef __attribute__((ext_vector_type(8)))  float    v8f;
typedef __attribute__((ext_vector_type(4)))  unsigned u32x4;
typedef __attribute__((ext_vector_type(8)))  int      i32x8;
typedef __attribute__((ext_vector_type(4)))  int      i32x4;

#define SEQ    4096
#define DMODEL 1024
#define NHEADS 16
#define HDIM   64

#if defined(__has_builtin)
#if __has_builtin(__builtin_amdgcn_tensor_load_to_lds)
#define HAVE_TDM 1
#endif
#endif

// ---------------------------------------------------------------------------
// WMMA wrapper: D = A(16x32 f16) * B(32x16 f16) + C(16x16 f32)
// ---------------------------------------------------------------------------
__device__ __forceinline__ v8f wmma_f16(v16h a, v16h b, v8f c) {
  return __builtin_amdgcn_wmma_f32_16x16x32_f16(
      /*neg_a=*/false, a, /*neg_b=*/false, b,
      /*c_mod=*/(short)0, c, /*reuse_a=*/false, /*reuse_b=*/false);
}

// ---------------------------------------------------------------------------
// Fragment loaders per CDNA5 ISA VGPR layouts (CDNA5 ISA §7.12.2).
//
// 16-bit A-matrix 16x32:
//   lanes 0-15 : row M=lane,     halves [0..7]=K 0..7,  [8..15]=K 16..23
//   lanes 16-31: row M=lane-16,  halves [0..7]=K 8..15, [8..15]=K 24..31
// B (32x16) from a row-major matrix whose ROWS are B's COLUMNS uses the same
// pattern (weight rows of W for y = x*W^T, key rows for Q*K^T).
// ---------------------------------------------------------------------------
__device__ __forceinline__ v16h load_frag_rm(const _Float16* __restrict__ base,
                                             int stride, int row0, int col0,
                                             int lane) {
  const int l  = lane & 15;
  const int hi = (lane >> 4) & 1;
  const _Float16* p = base + (size_t)(row0 + l) * (size_t)stride + col0 + hi * 8;
  v8h lo = *reinterpret_cast<const v8h*>(p);        // 16B -> *_load_b128
  v8h hh = *reinterpret_cast<const v8h*>(p + 16);
  v16h r;
#pragma unroll
  for (int i = 0; i < 8; ++i) { r[i] = lo[i]; r[8 + i] = hh[i]; }
  return r;
}

// B fragment from a genuinely row-major [K x N] matrix (e.g. V): column-strided.
__device__ __forceinline__ v16h load_frag_cm(const _Float16* __restrict__ base,
                                             int stride, int k0, int col0,
                                             int lane) {
  const int l  = lane & 15;
  const int hi = (lane >> 4) & 1;
  const _Float16* p = base + (size_t)(k0 + hi * 8) * (size_t)stride + col0 + l;
  v16h r;
#pragma unroll
  for (int i = 0; i < 8; ++i) r[i] = p[(size_t)i * stride];
  p += (size_t)16 * stride;
#pragma unroll
  for (int i = 0; i < 8; ++i) r[8 + i] = p[(size_t)i * stride];
  return r;
}

// C/D tile stores: lane n = col0 + (lane&15); VGPR v -> row0 + v + (lane>>4)*8
__device__ __forceinline__ void store_tile_f16(_Float16* __restrict__ dst,
                                               int stride, int row0, int col0,
                                               int lane, v8f c) {
  const int n  = col0 + (lane & 15);
  const int rb = row0 + ((lane >> 4) & 1) * 8;
#pragma unroll
  for (int vv = 0; vv < 8; ++vv)
    dst[(size_t)(rb + vv) * stride + n] = (_Float16)c[vv];
}

__device__ __forceinline__ void store_tile_f32(float* __restrict__ dst,
                                               int stride, int row0, int col0,
                                               int lane, v8f c) {
  const int n  = col0 + (lane & 15);
  const int rb = row0 + ((lane >> 4) & 1) * 8;
#pragma unroll
  for (int vv = 0; vv < 8; ++vv)
    dst[(size_t)(rb + vv) * stride + n] = c[vv];
}

// ---------------------------------------------------------------------------
// Tensor Data Mover: 2D f16 tile (64 x 64 elements) Global -> LDS.
// Builds the D# per CDNA5 ISA:
//   group0: count=1 | lds_addr | global_addr[56:0] | type=2
//   group1: data_size=2B, tensor_dim0/1, tile_dim0/1=64, tensor_dim0_stride
// ---------------------------------------------------------------------------
#ifdef HAVE_TDM
__device__ __forceinline__ void tdm_load_tile64_f16(_Float16* lds_dst,
                                                    const _Float16* gsrc,
                                                    unsigned row_len,
                                                    unsigned n_rows) {
  const unsigned lds_off = (unsigned)(size_t)(void*)lds_dst;
  const unsigned long long ga = (unsigned long long)(size_t)(const void*)gsrc;
  u32x4 g0;
  g0[0] = 1u;                                               // count=1 (valid)
  g0[1] = lds_off;                                          // LDS byte address
  g0[2] = (unsigned)ga;                                     // global_addr[31:0]
  g0[3] = (unsigned)((ga >> 32) & 0x1FFFFFFull) | (2u << 30); // [56:32] | type=2
  i32x8 g1;
  g1[0] = 0x00010000;                                       // data_size = 2 bytes
  g1[1] = (int)((row_len & 0xFFFFu) << 16);                 // tensor_dim0[15:0]
  g1[2] = (int)(((row_len >> 16) & 0xFFFFu) | ((n_rows & 0xFFFFu) << 16));
  g1[3] = (int)(((n_rows >> 16) & 0xFFFFu) | (64u << 16));  // tile_dim0 = 64
  g1[4] = 64;                                               // tile_dim1 = 64
  g1[5] = (int)row_len;                                     // dim0_stride[31:0]
  g1[6] = 0;
  g1[7] = 0;
  const i32x4 z4 = {0, 0, 0, 0};
#if defined(__clang_major__) && (__clang_major__ >= 23)
  const i32x8 z8 = {0, 0, 0, 0, 0, 0, 0, 0};
  __builtin_amdgcn_tensor_load_to_lds(g0, g1, z4, z4, z8, 0);
#else
  __builtin_amdgcn_tensor_load_to_lds(g0, g1, z4, z4, 0);
#endif
}

__device__ __forceinline__ void tdm_wait0() {
#if __has_builtin(__builtin_amdgcn_s_wait_tensorcnt)
  __builtin_amdgcn_s_wait_tensorcnt(0);
#else
  asm volatile("s_wait_tensorcnt 0x0" ::: "memory");
#endif
}
#endif  // HAVE_TDM

// ---------------------------------------------------------------------------
// [0] elementwise f32 -> f16
// ---------------------------------------------------------------------------
__global__ void cvt_f32_to_f16_kernel(const float* __restrict__ in,
                                      _Float16* __restrict__ out, int n) {
  int i = blockIdx.x * blockDim.x + threadIdx.x;
  if (i < n) out[i] = (_Float16)in[i];
}

// ---------------------------------------------------------------------------
// [1] fused QKV projection: Q/K/V[m,n] = sum_k x[m,k] * W{q,k,v}[n,k]
//     one wave per 16x16 tile; A fragment reused across the three B matrices.
// ---------------------------------------------------------------------------
__global__ void __launch_bounds__(128)
qkv_proj_kernel(const _Float16* __restrict__ xh,
                const _Float16* __restrict__ wq,
                const _Float16* __restrict__ wk,
                const _Float16* __restrict__ wv,
                _Float16* __restrict__ qo,
                _Float16* __restrict__ ko,
                _Float16* __restrict__ vo) {
  const int lane = threadIdx.x & 31;
  const int wave = threadIdx.x >> 5;
  const int tile = blockIdx.x * 4 + wave;   // 256*64 = 16384 tiles
  const int tm = tile >> 6;                 // /64
  const int tn = tile & 63;                 // %64
  const int row0 = tm * 16, col0 = tn * 16;

  v8f cq = {}, ck = {}, cv = {};
  for (int kk = 0; kk < DMODEL; kk += 32) {
    if (kk + 64 < DMODEL)
      __builtin_prefetch(xh + (size_t)row0 * DMODEL + kk + 64, 0, 3);
    v16h a  = load_frag_rm(xh, DMODEL, row0, kk, lane);
    v16h bq = load_frag_rm(wq, DMODEL, col0, kk, lane);
    v16h bk = load_frag_rm(wk, DMODEL, col0, kk, lane);
    v16h bv = load_frag_rm(wv, DMODEL, col0, kk, lane);
    cq = wmma_f16(a, bq, cq);
    ck = wmma_f16(a, bk, ck);
    cv = wmma_f16(a, bv, cv);
  }
  store_tile_f16(qo, DMODEL, row0, col0, lane, cq);
  store_tile_f16(ko, DMODEL, row0, col0, lane, ck);
  store_tile_f16(vo, DMODEL, row0, col0, lane, cv);
}

// ---------------------------------------------------------------------------
// [2] causal flash attention. grid = (S/64, H), block = 128 (4 waves).
//     Each wave: 16 query rows, full hd=64 accumulator, online softmax.
// ---------------------------------------------------------------------------
__global__ void __launch_bounds__(128)
flash_attn_kernel(const _Float16* __restrict__ qh,
                  const _Float16* __restrict__ kh,
                  const _Float16* __restrict__ vh,
                  _Float16* __restrict__ oh) {
  __shared__ _Float16 lds[4 * 16 * 32];     // per-wave 16x32 P tile
  const int lane   = threadIdx.x & 31;
  const int wave   = threadIdx.x >> 5;
  const int head   = blockIdx.y;
  const int q0     = blockIdx.x * 64 + wave * 16;
  const int c0     = head * HDIM;
  const int laneN  = lane & 15;
  const int halfId = (lane >> 4) & 1;
  const float scale = 0.125f;               // 1/sqrt(64)

  // Q fragments for the wave's 16 rows (hd = 2 x K32)
  const v16h aq0 = load_frag_rm(qh, DMODEL, q0, c0, lane);
  const v16h aq1 = load_frag_rm(qh, DMODEL, q0, c0 + 32, lane);

  v8f acc0 = {}, acc1 = {}, acc2 = {}, acc3 = {};
  float mrun[8], lrun[8];
#pragma unroll
  for (int i = 0; i < 8; ++i) { mrun[i] = -3.0e38f; lrun[i] = 0.0f; }

  _Float16* pl = lds + wave * (16 * 32);
  const int rowb = q0 + halfId * 8;

  for (int kb = 0; kb <= q0 + 15; kb += 32) {
    // ---- scores: two 16x16 key tiles, contraction over hd=64 ----
    v8f s0 = {}, s1 = {};
    {
      v16h b0 = load_frag_rm(kh, DMODEL, kb,      c0,      lane);
      v16h b1 = load_frag_rm(kh, DMODEL, kb,      c0 + 32, lane);
      s0 = wmma_f16(aq0, b0, s0);
      s0 = wmma_f16(aq1, b1, s0);
      v16h b2 = load_frag_rm(kh, DMODEL, kb + 16, c0,      lane);
      v16h b3 = load_frag_rm(kh, DMODEL, kb + 16, c0 + 32, lane);
      s1 = wmma_f16(aq0, b2, s1);
      s1 = wmma_f16(aq1, b3, s1);
    }

    const int key0 = kb + laneN;
    const int key1 = kb + 16 + laneN;
    float p0[8], p1[8], alpha[8];
#pragma unroll
    for (int vv = 0; vv < 8; ++vv) {
      const int row = rowb + vv;
      float a0 = (key0 <= row) ? s0[vv] * scale : -3.0e38f;   // causal mask
      float a1 = (key1 <= row) ? s1[vv] * scale : -3.0e38f;
      // row max across the 16 lanes of this half (row lives there)
      float mx = fmaxf(a0, a1);
      mx = fmaxf(mx, __shfl_xor(mx, 1));
      mx = fmaxf(mx, __shfl_xor(mx, 2));
      mx = fmaxf(mx, __shfl_xor(mx, 4));
      mx = fmaxf(mx, __shfl_xor(mx, 8));
      const float mnew = fmaxf(mrun[vv], mx);
      const float al   = __expf(mrun[vv] - mnew);
      mrun[vv]  = mnew;
      alpha[vv] = al;
      const float e0 = __expf(a0 - mnew);
      const float e1 = __expf(a1 - mnew);
      float rs = e0 + e1;
      rs += __shfl_xor(rs, 1);
      rs += __shfl_xor(rs, 2);
      rs += __shfl_xor(rs, 4);
      rs += __shfl_xor(rs, 8);
      lrun[vv] = lrun[vv] * al + rs;
      p0[vv] = e0;
      p1[vv] = e1;
    }
    // rescale running accumulator by alpha (per row)
#pragma unroll
    for (int vv = 0; vv < 8; ++vv) {
      acc0[vv] *= alpha[vv]; acc1[vv] *= alpha[vv];
      acc2[vv] *= alpha[vv]; acc3[vv] *= alpha[vv];
    }

    // ---- re-layout P: C-layout (lane=col) -> A fragment via LDS ----
#pragma unroll
    for (int vv = 0; vv < 8; ++vv) {
      const int lr = halfId * 8 + vv;
      pl[lr * 32 + laneN]      = (_Float16)p0[vv];
      pl[lr * 32 + 16 + laneN] = (_Float16)p1[vv];
    }
    asm volatile("s_wait_dscnt 0" ::: "memory");  // wave-local store->load order
    const v16h pa = load_frag_rm(pl, 32, 0, 0, lane);

    // ---- PV: contraction over the 32-key chunk, 4 hd column tiles ----
    v16h bv0 = load_frag_cm(vh, DMODEL, kb, c0,      lane);
    acc0 = wmma_f16(pa, bv0, acc0);
    v16h bv1 = load_frag_cm(vh, DMODEL, kb, c0 + 16, lane);
    acc1 = wmma_f16(pa, bv1, acc1);
    v16h bv2 = load_frag_cm(vh, DMODEL, kb, c0 + 32, lane);
    acc2 = wmma_f16(pa, bv2, acc2);
    v16h bv3 = load_frag_cm(vh, DMODEL, kb, c0 + 48, lane);
    acc3 = wmma_f16(pa, bv3, acc3);
  }

  // ---- normalize and store attention output (f16, [S, D] layout) ----
#pragma unroll
  for (int vv = 0; vv < 8; ++vv) {
    const float inv = 1.0f / lrun[vv];
    const int row = rowb + vv;
    _Float16* dst = oh + (size_t)row * DMODEL + c0 + laneN;
    dst[0]  = (_Float16)(acc0[vv] * inv);
    dst[16] = (_Float16)(acc1[vv] * inv);
    dst[32] = (_Float16)(acc2[vv] * inv);
    dst[48] = (_Float16)(acc3[vv] * inv);
  }
}

// ---------------------------------------------------------------------------
// [3] output projection: out[m,n] = sum_k attn[m,k] * Wo[n,k]  (f32 out)
//     TDM-staged: grid (64,16), block = 128 (4 waves) computes a 64x64 tile.
//     Per 64-K chunk: wave0 issues tensor_load_to_lds for A(64x64) and B(64x64)
//     tiles, waits TENSORcnt, then all waves run 8 WMMAs from LDS fragments.
// ---------------------------------------------------------------------------
__global__ void __launch_bounds__(128)
out_proj_kernel(const _Float16* __restrict__ ah,
                const _Float16* __restrict__ wo,
                float* __restrict__ out) {
  __shared__ _Float16 lA[64 * 64];   // 8 KB: rows row0..row0+63, K chunk
  __shared__ _Float16 lB[64 * 64];   // 8 KB: Wo rows col0..col0+63, K chunk
  const int lane = threadIdx.x & 31;
  const int wave = threadIdx.x >> 5;
  const int row0 = blockIdx.x * 64;
  const int col0 = blockIdx.y * 64;

  v8f c0 = {}, c1 = {}, c2 = {}, c3 = {};

  for (int k0 = 0; k0 < DMODEL; k0 += 64) {
#ifdef HAVE_TDM
    if (wave == 0) {
      tdm_load_tile64_f16(lA, ah + (size_t)row0 * DMODEL + k0, DMODEL, SEQ);
      tdm_load_tile64_f16(lB, wo + (size_t)col0 * DMODEL + k0, DMODEL, DMODEL);
      tdm_wait0();
    }
#else
    // fallback: cooperative b128 copy, 8 halves per thread per step
    for (int i = threadIdx.x; i < 64 * 8; i += 128) {
      const int r = i >> 3, s = (i & 7) * 8;
      *reinterpret_cast<v8h*>(lA + r * 64 + s) =
          *reinterpret_cast<const v8h*>(ah + (size_t)(row0 + r) * DMODEL + k0 + s);
      *reinterpret_cast<v8h*>(lB + r * 64 + s) =
          *reinterpret_cast<const v8h*>(wo + (size_t)(col0 + r) * DMODEL + k0 + s);
    }
#endif
    __syncthreads();
#pragma unroll
    for (int ks = 0; ks < 64; ks += 32) {
      v16h a  = load_frag_rm(lA, 64, wave * 16, ks, lane);
      v16h b0 = load_frag_rm(lB, 64, 0,  ks, lane);
      v16h b1 = load_frag_rm(lB, 64, 16, ks, lane);
      v16h b2 = load_frag_rm(lB, 64, 32, ks, lane);
      v16h b3 = load_frag_rm(lB, 64, 48, ks, lane);
      c0 = wmma_f16(a, b0, c0);
      c1 = wmma_f16(a, b1, c1);
      c2 = wmma_f16(a, b2, c2);
      c3 = wmma_f16(a, b3, c3);
    }
    __syncthreads();
  }

  store_tile_f32(out, DMODEL, row0 + wave * 16, col0,      lane, c0);
  store_tile_f32(out, DMODEL, row0 + wave * 16, col0 + 16, lane, c1);
  store_tile_f32(out, DMODEL, row0 + wave * 16, col0 + 32, lane, c2);
  store_tile_f32(out, DMODEL, row0 + wave * 16, col0 + 48, lane, c3);
}

// ---------------------------------------------------------------------------
// launcher
// ---------------------------------------------------------------------------
extern "C" void kernel_launch(void* const* d_in, const int* in_sizes, int n_in,
                              void* d_out, int out_size, void* d_ws, size_t ws_size,
                              hipStream_t stream) {
  const float* x  = (const float*)d_in[0];
  const float* Wq = (const float*)d_in[1];
  const float* Wk = (const float*)d_in[2];
  const float* Wv = (const float*)d_in[3];
  const float* Wo = (const float*)d_in[4];
  float* out = (float*)d_out;

  const size_t SX = (size_t)SEQ * DMODEL;      // 4M elements
  const size_t SW = (size_t)DMODEL * DMODEL;   // 1M elements

  _Float16* ws  = (_Float16*)d_ws;             // ~50 MB of f16 needed total
  _Float16* xh  = ws;
  _Float16* wqh = xh + SX;
  _Float16* wkh = wqh + SW;
  _Float16* wvh = wkh + SW;
  _Float16* woh = wvh + SW;
  _Float16* qh  = woh + SW;
  _Float16* kh  = qh + SX;
  _Float16* vh  = kh + SX;
  _Float16* ah  = vh + SX;

  // [0] f32 -> f16 conversions
  cvt_f32_to_f16_kernel<<<(int)((SX + 255) / 256), 256, 0, stream>>>(x,  xh,  (int)SX);
  cvt_f32_to_f16_kernel<<<(int)((SW + 255) / 256), 256, 0, stream>>>(Wq, wqh, (int)SW);
  cvt_f32_to_f16_kernel<<<(int)((SW + 255) / 256), 256, 0, stream>>>(Wk, wkh, (int)SW);
  cvt_f32_to_f16_kernel<<<(int)((SW + 255) / 256), 256, 0, stream>>>(Wv, wvh, (int)SW);
  cvt_f32_to_f16_kernel<<<(int)((SW + 255) / 256), 256, 0, stream>>>(Wo, woh, (int)SW);

  // [1] fused QKV projection: 16384 16x16 tiles, 4 waves / block
  qkv_proj_kernel<<<4096, 128, 0, stream>>>(xh, wqh, wkh, wvh, qh, kh, vh);

  // [2] causal flash attention: (64 query blocks) x (16 heads)
  flash_attn_kernel<<<dim3(64, 16), 128, 0, stream>>>(qh, kh, vh, ah);

  // [3] output projection -> f32 d_out (TDM-staged GEMM, 64x64 block tiles)
  out_proj_kernel<<<dim3(64, 16), 128, 0, stream>>>(ah, woh, out);
}
